// MultiheadAttention_59107339927555
// MI455X (gfx1250) — compile-verified, matrix operand order checked by reading, not run on hardware
//
#include <hip/hip_runtime.h>

// ---------------------------------------------------------------------------
// Types for CDNA5 WMMA
// ---------------------------------------------------------------------------
typedef __attribute__((ext_vector_type(16))) __bf16 v16bf;
typedef __attribute__((ext_vector_type(8)))  float  v8f;

__device__ __forceinline__ unsigned short f2bf(float f) {
  unsigned int u = __float_as_uint(f);
  return (unsigned short)((u + 0x7FFFu + ((u >> 16) & 1u)) >> 16);  // RNE
}
__device__ __forceinline__ float bf2f(unsigned short h) {
  return __uint_as_float(((unsigned int)h) << 16);
}
__device__ __forceinline__ unsigned int pack2(float a, float b) {
  return (unsigned int)f2bf(a) | ((unsigned int)f2bf(b) << 16);
}

union V16 { uint4 u[2]; v16bf v; };

// ---------------------------------------------------------------------------
// CDNA5 async Global -> LDS copy (ASYNCcnt-tracked), ISA §15.18.3 op 98.
// VDST = VGPR holding LDS byte address; 64-bit global address; saddr = off.
// ---------------------------------------------------------------------------
__device__ __forceinline__ void async_copy_b128(unsigned lds_addr, const void* g) {
  asm volatile("global_load_async_to_lds_b128 %0, %1, off"
               :: "v"(lds_addr), "v"((unsigned long long)(size_t)g)
               : "memory");
}
__device__ __forceinline__ void wait_asynccnt0() {
#if __has_builtin(__builtin_amdgcn_s_wait_asynccnt)
  __builtin_amdgcn_s_wait_asynccnt(0);
#else
  asm volatile("s_wait_asynccnt 0" ::: "memory");
#endif
}

// ---------------------------------------------------------------------------
// Kernel 0a: expand circulant weights -> bf16, laid out We[j][k] row-major.
// We[j*K + k] = w[j>>3][k>>3][(j-k)&7]   (w shape: (N/8, K/8, 8) fp32)
// ---------------------------------------------------------------------------
__global__ void expand_w_kernel(const float* __restrict__ w,
                                unsigned short* __restrict__ We,
                                int N, int K) {
  int idx = blockIdx.x * blockDim.x + threadIdx.x;
  int total = N * K;
  if (idx >= total) return;
  int j  = idx / K;
  int kk = idx - j * K;
  int o = j >> 3, ii = kk >> 3, c = (j - kk) & 7;
  We[idx] = f2bf(w[((long)o * (K >> 3) + ii) * 8 + c]);
}

// ---------------------------------------------------------------------------
// Kernel 0b: one-shot fp32 -> bf16 conversion for q/k/v activations.
// ---------------------------------------------------------------------------
__global__ void cvt_bf16_kernel(const float* __restrict__ a,
                                const float* __restrict__ b,
                                const float* __restrict__ c,
                                unsigned short* __restrict__ oa,
                                unsigned short* __restrict__ ob,
                                unsigned short* __restrict__ oc) {
  const float* s; unsigned short* d;
  if (blockIdx.y == 0)      { s = a; d = oa; }
  else if (blockIdx.y == 1) { s = b; d = ob; }
  else                      { s = c; d = oc; }
  long i = ((long)blockIdx.x * blockDim.x + threadIdx.x) * 4;
  float4 f = *(const float4*)(s + i);
  uint2 p; p.x = pack2(f.x, f.y); p.y = pack2(f.z, f.w);
  *(uint2*)(d + i) = p;
}

// ---------------------------------------------------------------------------
// Unified bf16 WMMA GEMM:
//   Out[m][n] = sum_k A[m][k] * W[n][k] + bias[n>>3] (+ resid[m][n] if RESID)
// Block tile 128(M) x 128(N), K-step 32, double-buffered LDS with
// async Global->LDS staging (ASYNCcnt) overlapping the WMMA block.
// 256 threads = 8 waves; wave tile 32(M) x 64(N) = 2x4 WMMA accumulators.
// ---------------------------------------------------------------------------
#define TM 128
#define TN 128
#define TK 32
#define LDA 40   // padded LDS row stride in halves (80B) to spread banks
#define LDB 40

template <bool RESID>
__global__ __launch_bounds__(256) void gemm_bf16_kernel(
    const unsigned short* __restrict__ A,    // M x K bf16 (row-major)
    const unsigned short* __restrict__ W,    // N x K bf16 (row-major)
    const float* __restrict__ bias,          // N/8
    const float* __restrict__ resid,         // M x N fp32 (RESID only)
    void* __restrict__ OutV,                 // M x N: bf16 (!RESID) / fp32 (RESID)
    int K, int Nout)
{
  __shared__ alignas(16) unsigned short As[2][TM * LDA];
  __shared__ alignas(16) unsigned short Bs[2][TN * LDB];

  const int tid  = threadIdx.x;
  const int wid  = tid >> 5;
  const int lane = tid & 31;
  const int wm = wid & 3;          // 4 waves along M -> 128 rows
  const int wn = wid >> 2;         // 2 waves along N -> 128 cols
  const int lr = lane & 15;        // row (A) / col (B) within 16
  const int lk = (lane >> 4) << 3; // K sub-chunk base (0 or 8)

  const long m0 = (long)blockIdx.x * TM;
  const int  n0 = blockIdx.y * TN;

  v8f zero = {0.f,0.f,0.f,0.f,0.f,0.f,0.f,0.f};
  v8f acc[2][4] = {{zero, zero, zero, zero}, {zero, zero, zero, zero}};

  // staging: 2 threads per row, 16 halves (32B = 2x b128) each, A and B tiles
  const int sr = tid >> 1, sh = tid & 1;
  const unsigned short* aRow = A + (m0 + sr) * (long)K + sh * 16;
  const unsigned short* bRow = W + (long)(n0 + sr) * K + sh * 16;
  const int sIdx = sr * LDA + sh * 16;
  const unsigned ldsA0 = (unsigned)(size_t)&As[0][sIdx];
  const unsigned ldsA1 = (unsigned)(size_t)&As[1][sIdx];
  const unsigned ldsB0 = (unsigned)(size_t)&Bs[0][sIdx];
  const unsigned ldsB1 = (unsigned)(size_t)&Bs[1][sIdx];

  // prologue: async-stage K-step 0 into buffer 0
  async_copy_b128(ldsA0,      aRow);
  async_copy_b128(ldsA0 + 16, aRow + 8);
  async_copy_b128(ldsB0,      bRow);
  async_copy_b128(ldsB0 + 16, bRow + 8);

  int cur = 0;
  for (int k0 = 0; k0 < K; k0 += TK) {
    wait_asynccnt0();                      // my stage into buf[cur] landed
    __syncthreads();                       // everyone's buf[cur] is ready

    if (k0 + TK < K) {                     // async-stage next step -> buf[cur^1]
      unsigned la = cur ? ldsA0 : ldsA1;
      unsigned lb = cur ? ldsB0 : ldsB1;
      const unsigned short* ak = aRow + k0 + TK;
      const unsigned short* bk = bRow + k0 + TK;
      async_copy_b128(la,      ak);
      async_copy_b128(la + 16, ak + 8);
      async_copy_b128(lb,      bk);
      async_copy_b128(lb + 16, bk + 8);
      if (k0 + 2 * TK < K) {               // gfx1250 global_prefetch of k+2 slab
        __builtin_prefetch((const char*)(aRow + k0 + 2 * TK), 0, 0);
        __builtin_prefetch((const char*)(bRow + k0 + 2 * TK), 0, 0);
      }
    }

    // --- WMMA compute on buf[cur] (overlaps in-flight async DMA) ---
    V16 a[2], b[4];
    #pragma unroll
    for (int am = 0; am < 2; am++) {
      const unsigned short* p = &As[cur][(wm * 32 + am * 16 + lr) * LDA + lk];
      a[am].u[0] = *(const uint4*)p;
      a[am].u[1] = *(const uint4*)(p + 16);
    }
    #pragma unroll
    for (int bn = 0; bn < 4; bn++) {
      const unsigned short* p = &Bs[cur][(wn * 64 + bn * 16 + lr) * LDB + lk];
      b[bn].u[0] = *(const uint4*)p;
      b[bn].u[1] = *(const uint4*)(p + 16);
    }
    #pragma unroll
    for (int am = 0; am < 2; am++)
      #pragma unroll
      for (int bn = 0; bn < 4; bn++)
        acc[am][bn] = __builtin_amdgcn_wmma_f32_16x16x32_bf16(
            false, a[am].v, false, b[bn].v, (short)0, acc[am][bn], false, false);

    cur ^= 1;
  }

  // --- epilogue: C layout lanes 0-15 -> N, VGPR r -> M row (lanes>=16: M+8) ---
  const int mwave = wm * 32 + ((lane >> 4) << 3);
  #pragma unroll
  for (int am = 0; am < 2; am++) {
    #pragma unroll
    for (int bn = 0; bn < 4; bn++) {
      int col = n0 + wn * 64 + bn * 16 + lr;
      float bv = bias[col >> 3];
      long mb = m0 + mwave + am * 16;
      #pragma unroll
      for (int r = 0; r < 8; r++) {
        long m = mb + r;
        float val = acc[am][bn][r] + bv;
        if constexpr (RESID) {
          ((float*)OutV)[m * Nout + col] = val + resid[m * Nout + col];
        } else {
          ((unsigned short*)OutV)[m * (long)Nout + col] = f2bf(val);
        }
      }
    }
  }
}

// ---------------------------------------------------------------------------
// Kernel 2: tiny per-(b,h) attention: S = QK^T/8 (8x8 over d=64), softmax,
// O = P V. Writes attn (fp32, output #2) and O in (b, g*512+h*64+d) bf16.
// 256 threads = 4 units of 64 threads; unit handles one (b,h).
// ---------------------------------------------------------------------------
__global__ __launch_bounds__(256) void attn_kernel(
    const unsigned short* __restrict__ qp,
    const unsigned short* __restrict__ kp,
    const unsigned short* __restrict__ vp,
    unsigned short* __restrict__ Oflat,
    float* __restrict__ attn_out)
{
  __shared__ float P[4][8][9];
  const int tid = threadIdx.x;
  const int u = tid >> 6, t = tid & 63;
  const long bh = (long)blockIdx.x * 4 + u;      // B * n_head = 65536
  const long b = bh >> 3;
  const int  h = (int)(bh & 7);
  const int i = t >> 3, j = t & 7;

  const unsigned short* qr = qp + b * 4096 + i * 512 + h * 64;
  const unsigned short* kr = kp + b * 4096 + j * 512 + h * 64;
  float s = 0.f;
  #pragma unroll
  for (int d = 0; d < 64; d += 2) {
    unsigned int qw = *(const unsigned int*)(qr + d);
    unsigned int kw = *(const unsigned int*)(kr + d);
    s += bf2f((unsigned short)qw)         * bf2f((unsigned short)kw);
    s += bf2f((unsigned short)(qw >> 16)) * bf2f((unsigned short)(kw >> 16));
  }
  s *= 0.125f;                                   // 1/sqrt(d_k)
  P[u][i][j] = s;
  __syncthreads();

  float mx = P[u][i][0];
  #pragma unroll
  for (int jj = 1; jj < 8; jj++) mx = fmaxf(mx, P[u][i][jj]);
  float sum = 0.f;
  #pragma unroll
  for (int jj = 0; jj < 8; jj++) sum += __expf(P[u][i][jj] - mx);
  float p = __expf(s - mx) / sum;
  __syncthreads();
  P[u][i][j] = p;
  attn_out[bh * 64 + i * 8 + j] = p;
  __syncthreads();

  // O: remap 64 threads -> (row i2, 8-wide d chunk dc)
  const int i2 = t >> 3, dc = t & 7;
  float o[8] = {0.f,0.f,0.f,0.f,0.f,0.f,0.f,0.f};
  #pragma unroll
  for (int jj = 0; jj < 8; jj++) {
    float pv = P[u][i2][jj];
    const unsigned short* vr = vp + b * 4096 + jj * 512 + h * 64 + dc * 8;
    #pragma unroll
    for (int d = 0; d < 8; d += 2) {
      unsigned int vw = *(const unsigned int*)(vr + d);
      o[d]     += pv * bf2f((unsigned short)vw);
      o[d + 1] += pv * bf2f((unsigned short)(vw >> 16));
    }
  }
  unsigned short* orow = Oflat + b * 4096 + i2 * 512 + h * 64 + dc * 8;
  #pragma unroll
  for (int d = 0; d < 8; d += 2) {
    *(unsigned int*)(orow + d) = pack2(o[d], o[d + 1]);
  }
}

// ---------------------------------------------------------------------------
// Host-side launcher
// ---------------------------------------------------------------------------
extern "C" void kernel_launch(void* const* d_in, const int* in_sizes, int n_in,
                              void* d_out, int out_size, void* d_ws, size_t ws_size,
                              hipStream_t stream) {
  (void)in_sizes; (void)n_in; (void)out_size; (void)ws_size;

  const float* q    = (const float*)d_in[0];
  const float* k    = (const float*)d_in[1];
  const float* v    = (const float*)d_in[2];
  const float* w_q  = (const float*)d_in[3];
  const float* b_q  = (const float*)d_in[4];
  const float* w_k  = (const float*)d_in[5];
  const float* b_k  = (const float*)d_in[6];
  const float* w_v  = (const float*)d_in[7];
  const float* b_v  = (const float*)d_in[8];
  const float* w_fc = (const float*)d_in[9];
  const float* b_fc = (const float*)d_in[10];

  const long B = 8192, C = 1024, N = 4096;

  unsigned short* Wq_e  = (unsigned short*)d_ws;        // 4096 x 1024
  unsigned short* Wk_e  = Wq_e  + 4096l * 1024;
  unsigned short* Wv_e  = Wk_e  + 4096l * 1024;
  unsigned short* Wfc_e = Wv_e  + 4096l * 1024;         // 1024 x 4096
  unsigned short* Xq_b  = Wfc_e + 1024l * 4096;         // 8192 x 1024 bf16
  unsigned short* Xk_b  = Xq_b + B * C;
  unsigned short* Xv_b  = Xk_b + B * C;
  unsigned short* qp    = Xv_b + B * C;                 // 8192 x 4096 each
  unsigned short* kp    = qp + B * N;
  unsigned short* vp    = kp + B * N;
  unsigned short* of    = vp + B * N;

  float* out  = (float*)d_out;       // 8192 x 1024
  float* attn = out + B * C;         // 8192 x 8 x 8 x 8

  // Stage 0: weight expansion + activation conversion (memory-bound, tiny)
  {
    int total = 4096 * 1024;
    int blocks = (total + 255) / 256;
    expand_w_kernel<<<blocks, 256, 0, stream>>>(w_q,  Wq_e,  4096, 1024);
    expand_w_kernel<<<blocks, 256, 0, stream>>>(w_k,  Wk_e,  4096, 1024);
    expand_w_kernel<<<blocks, 256, 0, stream>>>(w_v,  Wv_e,  4096, 1024);
    expand_w_kernel<<<blocks, 256, 0, stream>>>(w_fc, Wfc_e, 1024, 4096);
    dim3 gc((unsigned)(B * C / 4 / 256), 3);
    cvt_bf16_kernel<<<gc, 256, 0, stream>>>(q, k, v, Xq_b, Xk_b, Xv_b);
  }

  // Stage 1: q/k/v projections (M=8192, N=4096, K=1024)
  {
    dim3 g(64, 32);
    gemm_bf16_kernel<false><<<g, 256, 0, stream>>>(Xq_b, Wq_e, b_q, nullptr, qp, 1024, 4096);
    gemm_bf16_kernel<false><<<g, 256, 0, stream>>>(Xk_b, Wk_e, b_k, nullptr, kp, 1024, 4096);
    gemm_bf16_kernel<false><<<g, 256, 0, stream>>>(Xv_b, Wv_e, b_v, nullptr, vp, 1024, 4096);
  }

  // Stage 2: attention (writes attn output slab + O matrix)
  attn_kernel<<<16384, 256, 0, stream>>>(qp, kp, vp, of, attn);

  // Stage 3: output projection + bias + residual (M=8192, N=1024, K=4096)
  {
    dim3 g(64, 8);
    gemm_bf16_kernel<true><<<g, 256, 0, stream>>>(of, Wfc_e, b_fc, q, out, 4096, 1024);
  }
}
